// MatchLSTM_67645734912550
// MI455X (gfx1250) — matrix operand order repaired and checked by reading, not compiled, not run on hardware
//
#include <hip/hip_runtime.h>
#include <cmath>

typedef __bf16 bf16_t;
typedef __attribute__((ext_vector_type(16))) __bf16 v16bf;
typedef __attribute__((ext_vector_type(8)))  __bf16 v8bf;
typedef __attribute__((ext_vector_type(8)))  float  v8f;

#define Hd  300   // hidden
#define Bb  32    // batch
#define Pp  256   // premise length
#define Tt  128   // time steps
#define HP  320   // H padded to mult of 32
#define K2  640   // [x | h] padded K
#define K3  960   // [a | x | h] padded K
#define N4  1200  // 4H (already mult of 16)

__device__ __forceinline__ float sigm(float x) { return 1.0f / (1.0f + __expf(-x)); }

// ---------------- staging / conversion kernels ----------------

__global__ void k_prep_premise(const float* __restrict__ prem, bf16_t* __restrict__ out) {
    int idx = blockIdx.x * blockDim.x + threadIdx.x;
    if (idx >= Bb * Pp * HP) return;
    int row = idx / HP, k = idx % HP;
    out[idx] = (k < Hd) ? (bf16_t)prem[row * Hd + k] : (bf16_t)0.0f;
}

// wsT[n][k] = ws[k][n], zero-padded to [HP,HP]
__global__ void k_prep_ws(const float* __restrict__ w, bf16_t* __restrict__ out) {
    int idx = blockIdx.x * blockDim.x + threadIdx.x;
    if (idx >= HP * HP) return;
    int n = idx / HP, k = idx % HP;
    out[idx] = (n < Hd && k < Hd) ? (bf16_t)w[k * Hd + n] : (bf16_t)0.0f;
}

// wtwmT[n][k]: k in [0,HP) -> wt[k][n], k in [HP,2HP) -> wm[k-HP][n]
__global__ void k_prep_wtwm(const float* __restrict__ wt, const float* __restrict__ wm,
                            bf16_t* __restrict__ out) {
    int idx = blockIdx.x * blockDim.x + threadIdx.x;
    if (idx >= HP * K2) return;
    int n = idx / K2, k = idx % K2;
    float v = 0.0f;
    if (n < Hd) {
        if (k < HP) { if (k < Hd) v = wt[k * Hd + n]; }
        else        { int kk = k - HP; if (kk < Hd) v = wm[kk * Hd + n]; }
    }
    out[idx] = (bf16_t)v;
}

// kernT[n][k]: k = sec*HP + kk, sec in {a,x,h}; -> kernel[(sec*H+kk)*4H + n]
__global__ void k_prep_kernel(const float* __restrict__ kern, bf16_t* __restrict__ out) {
    int idx = blockIdx.x * blockDim.x + threadIdx.x;
    if (idx >= N4 * K3) return;
    int n = idx / K3, k = idx % K3;
    int sec = k / HP, kk = k % HP;
    float v = (kk < Hd) ? kern[(sec * Hd + kk) * N4 + n] : 0.0f;
    out[idx] = (bf16_t)v;
}

__global__ void k_init(bf16_t* __restrict__ abf, float* __restrict__ cbuf) {
    int idx = blockIdx.x * blockDim.x + threadIdx.x;
    if (idx < Bb * K3) abf[idx] = (bf16_t)0.0f;
    if (idx < Bb * HP) cbuf[idx] = 0.0f;
}

__global__ void k_stage_x(const float* __restrict__ hyp, bf16_t* __restrict__ abf, int t) {
    int idx = blockIdx.x * blockDim.x + threadIdx.x;
    if (idx >= Bb * HP) return;
    int b = idx / HP, k = idx % HP;
    abf[b * K3 + HP + k] = (k < Hd) ? (bf16_t)hyp[(b * Tt + t) * Hd + k] : (bf16_t)0.0f;
}

// ---------------- bf16 WMMA GEMM: C[M,N] = A[M,K] * BT[N,K]^T ----------------
// One wave per 16x16 tile. A row-major (lda, + column offset), BT row-major [N,K].
__global__ void k_gemm_bf16(const bf16_t* __restrict__ A, const bf16_t* __restrict__ BT,
                            float* __restrict__ C,
                            int M, int N, int K, int lda, int ldb, int ldc, int aoff) {
    int wpb = blockDim.x >> 5;
    int wid = blockIdx.x * wpb + (threadIdx.x >> 5);
    int ntn = N >> 4;
    if (wid >= (M >> 4) * ntn) return;            // whole-wave exit: EXEC stays all-1s
    int tm = wid / ntn, tn = wid % ntn;
    int lane = threadIdx.x & 31;
    int l15  = lane & 15;
    int hi   = lane >> 4;

    const bf16_t* arow = A + (size_t)(tm * 16 + l15) * lda + aoff;
    const bf16_t* brow = BT + (size_t)(tn * 16 + l15) * ldb;

    v8f acc = {0.f, 0.f, 0.f, 0.f, 0.f, 0.f, 0.f, 0.f};
    for (int k0 = 0; k0 < K; k0 += 32) {
        const bf16_t* ap = arow + k0 + hi * 8;
        v8bf alo = *(const v8bf*)(ap);
        v8bf ahi = *(const v8bf*)(ap + 16);
        const bf16_t* bp = brow + k0 + hi * 8;
        v8bf blo = *(const v8bf*)(bp);
        v8bf bhi = *(const v8bf*)(bp + 16);
        v16bf av = __builtin_shufflevector(alo, ahi, 0,1,2,3,4,5,6,7,8,9,10,11,12,13,14,15);
        v16bf bv = __builtin_shufflevector(blo, bhi, 0,1,2,3,4,5,6,7,8,9,10,11,12,13,14,15);
        acc = __builtin_amdgcn_wmma_f32_16x16x32_bf16(false, av, false, bv,
                                                      (short)0, acc, false, false);
    }
    float* crow = C + (size_t)(tm * 16 + hi * 8) * ldc + tn * 16 + l15;
#pragma unroll
    for (int r = 0; r < 8; ++r) crow[(size_t)r * ldc] = acc[r];
}

// ---- ds GEMM variant: C stored transposed per batch: dsT[b][n][p], p = row % Pp ----
// M = Bb*Pp rows (tile never straddles a batch boundary since 256 % 16 == 0).
__global__ void k_gemm_bf16_dsT(const bf16_t* __restrict__ A, const bf16_t* __restrict__ BT,
                                float* __restrict__ dsT, int N, int K, int lda, int ldb) {
    int wpb = blockDim.x >> 5;
    int wid = blockIdx.x * wpb + (threadIdx.x >> 5);
    int ntn = N >> 4;
    if (wid >= ((Bb * Pp) >> 4) * ntn) return;
    int tm = wid / ntn, tn = wid % ntn;
    int lane = threadIdx.x & 31;
    int l15  = lane & 15;
    int hi   = lane >> 4;

    const bf16_t* arow = A + (size_t)(tm * 16 + l15) * lda;
    const bf16_t* brow = BT + (size_t)(tn * 16 + l15) * ldb;

    v8f acc = {0.f, 0.f, 0.f, 0.f, 0.f, 0.f, 0.f, 0.f};
    for (int k0 = 0; k0 < K; k0 += 32) {
        const bf16_t* ap = arow + k0 + hi * 8;
        v8bf alo = *(const v8bf*)(ap);
        v8bf ahi = *(const v8bf*)(ap + 16);
        const bf16_t* bp = brow + k0 + hi * 8;
        v8bf blo = *(const v8bf*)(bp);
        v8bf bhi = *(const v8bf*)(bp + 16);
        v16bf av = __builtin_shufflevector(alo, ahi, 0,1,2,3,4,5,6,7,8,9,10,11,12,13,14,15);
        v16bf bv = __builtin_shufflevector(blo, bhi, 0,1,2,3,4,5,6,7,8,9,10,11,12,13,14,15);
        acc = __builtin_amdgcn_wmma_f32_16x16x32_bf16(false, av, false, bv,
                                                      (short)0, acc, false, false);
    }
    // D elem r: row m = tm*16 + hi*8 + r, col n = tn*16 + l15.
    // Store at dsT[b*HP*Pp + n*Pp + p], b = m/Pp, p = m%Pp -> contiguous in r.
    int m0 = tm * 16 + hi * 8;
    int b  = m0 / Pp;
    int p0 = m0 % Pp;
    int n  = tn * 16 + l15;
    float* crow = dsT + (size_t)b * HP * Pp + (size_t)n * Pp + p0;
#pragma unroll
    for (int r = 0; r < 8; ++r) crow[r] = acc[r];
}

// ---------------- attention: energies -> softmax -> weighted premise sum ----------------
__global__ void k_attn(const float* __restrict__ sbuf,  // [B,HP] = dt+dm
                       const float* __restrict__ dsT,   // [B,HP,Pp] transposed
                       const float* __restrict__ we,    // [H]
                       const float* __restrict__ pmask, // [B*P]
                       const float* __restrict__ prem,  // [B,P,H]
                       bf16_t* __restrict__ abf) {      // write a_k to abf[:,0:HP]
    int b = blockIdx.x;
    int tid = threadIdx.x;  // == p, blockDim == Pp
    __shared__ float sh_s[HP];
    __shared__ float sh_we[Hd];
    __shared__ float red[Pp];
    __shared__ float alpha[Pp];

    for (int i = tid; i < HP; i += Pp) sh_s[i] = sbuf[b * HP + i];
    for (int i = tid; i < Hd; i += Pp) sh_we[i] = we[i];
    __syncthreads();

    const float* dsb = dsT + (size_t)b * HP * Pp + tid;  // coalesced across lanes at each h
    float e = 0.0f;
    for (int h = 0; h < Hd; ++h) e += tanhf(sh_s[h] + dsb[(size_t)h * Pp]) * sh_we[h];
    e += (1.0f - pmask[b * Pp + tid]) * -3.40282347e38f;

    red[tid] = e; __syncthreads();
    for (int s = Pp / 2; s > 0; s >>= 1) {
        if (tid < s) red[tid] = fmaxf(red[tid], red[tid + s]);
        __syncthreads();
    }
    float mx = red[0]; __syncthreads();
    float ex = __expf(e - mx);
    red[tid] = ex; __syncthreads();
    for (int s = Pp / 2; s > 0; s >>= 1) {
        if (tid < s) red[tid] += red[tid + s];
        __syncthreads();
    }
    float inv = 1.0f / red[0]; __syncthreads();
    alpha[tid] = ex * inv; __syncthreads();

    for (int h = tid; h < HP; h += Pp) {
        float acc = 0.0f;
        if (h < Hd) {
            const float* pb = prem + (size_t)b * Pp * Hd + h;
            for (int p = 0; p < Pp; ++p) acc += alpha[p] * pb[(size_t)p * Hd];
        }
        abf[b * K3 + h] = (bf16_t)acc;  // h>=Hd stays 0 (matches zeroed weight rows)
    }
}

// ---------- LSTM gate nonlinearity + state update; also stages x_{t+1} ----------
__global__ void k_lstm(const float* __restrict__ gates, const float* __restrict__ bias,
                       float* __restrict__ cbuf, bf16_t* __restrict__ abf,
                       const float* __restrict__ hyp, float* __restrict__ out, int t) {
    int idx = blockIdx.x * blockDim.x + threadIdx.x;
    if (idx >= Bb * Hd) return;
    int b = idx / Hd, h = idx % Hd;
    const float* g = gates + (size_t)b * N4;
    float ig = g[h]          + bias[h];
    float jg = g[Hd + h]     + bias[Hd + h];
    float fg = g[2 * Hd + h] + bias[2 * Hd + h];
    float og = g[3 * Hd + h] + bias[3 * Hd + h];
    float c  = cbuf[b * HP + h];
    float nc = c * sigm(fg + 1.0f) + sigm(ig) * tanhf(jg);
    float nh = tanhf(nc) * sigm(og);
    cbuf[b * HP + h] = nc;
    abf[b * K3 + 2 * HP + h] = (bf16_t)nh;            // h feeds next step's GEMMs
    out[((size_t)b * Tt + t) * Hd + h] = nh;
    if (t + 1 < Tt)                                   // stage x_{t+1} (padding stays 0)
        abf[b * K3 + HP + h] = (bf16_t)hyp[((size_t)b * Tt + t + 1) * Hd + h];
}

// ---------------- host-side orchestration ----------------
extern "C" void kernel_launch(void* const* d_in, const int* in_sizes, int n_in,
                              void* d_out, int out_size, void* d_ws, size_t ws_size,
                              hipStream_t stream) {
    const float* premise = (const float*)d_in[0];
    const float* pmask   = (const float*)d_in[1];
    const float* hyp     = (const float*)d_in[2];
    const float* ws_w    = (const float*)d_in[3];
    const float* wt_w    = (const float*)d_in[4];
    const float* wm_w    = (const float*)d_in[5];
    const float* we_w    = (const float*)d_in[6];
    const float* kern    = (const float*)d_in[7];
    const float* bias    = (const float*)d_in[8];
    float* out = (float*)d_out;

    char* base = (char*)d_ws;
    size_t off = 0;
    auto carve = [&](size_t bytes) -> void* {
        void* p = base + off;
        off += (bytes + 255) & ~(size_t)255;
        return p;
    };
    bf16_t* premise_bf = (bf16_t*)carve((size_t)Bb * Pp * HP * 2);
    bf16_t* wsT        = (bf16_t*)carve((size_t)HP * HP * 2);
    bf16_t* wtwmT      = (bf16_t*)carve((size_t)HP * K2 * 2);
    bf16_t* kernT      = (bf16_t*)carve((size_t)N4 * K3 * 2);
    float*  dsT        = (float*) carve((size_t)Bb * HP * Pp * 4);
    float*  sbuf       = (float*) carve((size_t)Bb * HP * 4);
    bf16_t* abf        = (bf16_t*)carve((size_t)Bb * K3 * 2);
    float*  gates      = (float*) carve((size_t)Bb * N4 * 4);
    float*  cbuf       = (float*) carve((size_t)Bb * HP * 4);
    (void)ws_size; (void)in_sizes; (void)n_in; (void)out_size;

    const int th = 256;
    k_prep_premise<<<(Bb * Pp * HP + th - 1) / th, th, 0, stream>>>(premise, premise_bf);
    k_prep_ws<<<(HP * HP + th - 1) / th, th, 0, stream>>>(ws_w, wsT);
    k_prep_wtwm<<<(HP * K2 + th - 1) / th, th, 0, stream>>>(wt_w, wm_w, wtwmT);
    k_prep_kernel<<<(N4 * K3 + th - 1) / th, th, 0, stream>>>(kern, kernT);
    k_init<<<(Bb * K3 + th - 1) / th, th, 0, stream>>>(abf, cbuf);
    k_stage_x<<<(Bb * HP + th - 1) / th, th, 0, stream>>>(hyp, abf, 0);

    // dsT = (premise @ ws) stored [B,H,P] : [8192,320] x [320,320]
    {
        int nt = ((Bb * Pp) / 16) * (HP / 16);
        k_gemm_bf16_dsT<<<(nt + 3) / 4, 128, 0, stream>>>(premise_bf, wsT, dsT,
                                                          HP, HP, HP, HP);
    }

    for (int t = 0; t < Tt; ++t) {
        // s = dt + dm = [x|h] @ [wt;wm] : [32,640] x [640,320]
        {
            int nt = (Bb / 16) * (HP / 16);
            k_gemm_bf16<<<(nt + 3) / 4, 128, 0, stream>>>(abf, wtwmT, sbuf,
                                                          Bb, HP, K2, K3, K2, HP, HP);
        }
        k_attn<<<Bb, Pp, 0, stream>>>(sbuf, dsT, we_w, pmask, premise, abf);
        // gates = [a|x|h] @ kernel : [32,960] x [960,1200]
        {
            int nt = (Bb / 16) * (N4 / 16);
            k_gemm_bf16<<<(nt + 3) / 4, 128, 0, stream>>>(abf, kernT, gates,
                                                          Bb, N4, K3, K3, K3, N4, 0);
        }
        k_lstm<<<(Bb * Hd + th - 1) / th, th, 0, stream>>>(gates, bias, cbuf, abf,
                                                           hyp, out, t);
    }
}
